// SSIMLoss_10763188044098
// MI455X (gfx1250) — matrix operand order repaired and checked by reading, not compiled, not run on hardware
//
#include <hip/hip_runtime.h>

typedef __attribute__((ext_vector_type(16))) _Float16 v16h;
typedef __attribute__((ext_vector_type(8)))  _Float16 v8h;
typedef __attribute__((ext_vector_type(8)))  float    v8f;

// Unaligned (4-byte) 8-float vector load wrapper: ROCm configures the memory
// system for unaligned access, so the backend can emit wide global loads at
// dword alignment.
struct __attribute__((packed, aligned(4))) V8FU { v8f v; };

#define IMG_H 512
#define IMG_W 512
#define NPLANE 64
#define TILES_X (IMG_W/16)                 // 32
#define TILES_PER_PLANE (TILES_X*TILES_X)  // 1024
#define NTILES (NPLANE*TILES_PER_PLANE)    // 65536
#define WAVES_PER_BLOCK 4
#define NBLOCKS (NTILES/WAVES_PER_BLOCK)   // 16384

// LDS staging strides (in _Float16 units). 40 halfs = 80 B per column keeps
// every 8-half run 16-byte aligned (b128 LDS traffic).
#define COL_STRIDE 40
#define SIG_STRIDE (16*COL_STRIDE)         // 640
#define WAVE_STRIDE (5*SIG_STRIDE)         // 3200

// Gaussian(sigma=1.5), 11 taps, normalized (matches the reference window).
__constant__ float c_G[11] = {
  0.00102838f, 0.00759876f, 0.03600077f, 0.10936069f, 0.21300553f,
  0.26601172f,
  0.21300553f, 0.10936069f, 0.03600077f, 0.00759876f, 0.00102838f
};

__global__ __launch_bounds__(128) void ssim_main_kernel(
    const float* __restrict__ img1, const float* __restrict__ img2,
    float* __restrict__ blocksum)
{
  __shared__ alignas(16) _Float16 sh[WAVES_PER_BLOCK * WAVE_STRIDE];
  __shared__ float sred[128];

  const int tid  = threadIdx.x;
  const int wave = tid >> 5;
  const int lane = tid & 31;
  const int hs   = lane >> 4;   // which 16-lane half
  const int lrow = lane & 15;

  // one 16x16 output tile per wave
  const int t   = blockIdx.x * WAVES_PER_BLOCK + wave;
  const int pl  = t >> 10;
  const int tin = t & 1023;
  const int tr0 = (tin >> 5) << 4;   // tile row origin
  const int tc0 = (tin & 31) << 4;   // tile col origin

  const float* p1 = img1 + (size_t)pl * (IMG_H * IMG_W);
  const float* p2 = img2 + (size_t)pl * (IMG_H * IMG_W);

  // Wave-uniform: does this tile's 32x32 input footprint stay in-bounds?
  const bool interior = (tr0 >= 5) && (tr0 + 26 < IMG_H) &&
                        (tc0 >= 5) && (tc0 + 26 < IMG_W);

  // Constant banded Gaussian operands.
  // Column-pass A (16x32, MxK): A[M][K] = g[K-M], K-M in [0,10]
  // Row-pass    B (32x16, KxN): B[K][N] = g[K-N], K-N in [0,10]
  v16h aG, bG;
  #pragma unroll
  for (int e = 0; e < 16; ++e) {
    int Ka = hs * 16 + e;                                      // A layout
    int da = Ka - lrow;
    aG[e] = (da >= 0 && da <= 10) ? (_Float16)c_G[da] : (_Float16)0.0f;
    int Kb = (e < 8) ? (hs * 8 + e) : (16 + hs * 8 + (e - 8)); // B layout
    int db = Kb - lrow;
    bG[e] = (db >= 0 && db <= 10) ? (_Float16)c_G[db] : (_Float16)0.0f;
  }

  _Float16* myLds = &sh[wave * WAVE_STRIDE];

  // ---------------- row pass: 2 chunks x 5 signals, one WMMA each ----------
  #pragma unroll
  for (int chunk = 0; chunk < 2; ++chunk) {
    const int  r    = tr0 - 5 + chunk * 16 + lrow;   // my A-matrix row M=lrow
    const int  roff = r * IMG_W;
    float xv[16], yv[16];
    if (interior) {
      // Fast path (~88% of tiles): two contiguous 8-float runs per image.
      const float* b1 = p1 + roff + (tc0 - 5);
      const float* b2 = p2 + roff + (tc0 - 5);
      v8f x0 = ((const V8FU*)(b1 + hs * 8))->v;
      v8f x1 = ((const V8FU*)(b1 + 16 + hs * 8))->v;
      v8f y0 = ((const V8FU*)(b2 + hs * 8))->v;
      v8f y1 = ((const V8FU*)(b2 + 16 + hs * 8))->v;
      #pragma unroll
      for (int i = 0; i < 8; ++i) {
        xv[i] = x0[i]; xv[8 + i] = x1[i];
        yv[i] = y0[i]; yv[8 + i] = y1[i];
      }
    } else {
      // Edge path: SAME zero padding, per-element guards.
      const bool rok = (r >= 0) && (r < IMG_H);
      #pragma unroll
      for (int e = 0; e < 16; ++e) {
        int K = (e < 8) ? (hs * 8 + e) : (16 + hs * 8 + (e - 8));
        int c = tc0 - 5 + K;
        bool ok = rok && (c >= 0) && (c < IMG_W);
        xv[e] = ok ? p1[roff + c] : 0.0f;
        yv[e] = ok ? p2[roff + c] : 0.0f;
      }
    }
    #pragma unroll
    for (int sig = 0; sig < 5; ++sig) {
      v16h a;
      #pragma unroll
      for (int e = 0; e < 16; ++e) {
        float v;
        switch (sig) {
          case 0:  v = xv[e];         break;  // x   -> mu1
          case 1:  v = yv[e];         break;  // y   -> mu2
          case 2:  v = xv[e] * xv[e]; break;  // x^2
          case 3:  v = yv[e] * yv[e]; break;  // y^2
          default: v = xv[e] * yv[e]; break;  // x*y
        }
        a[e] = (_Float16)v;
      }
      v8f cz = {};
      v8f d = __builtin_amdgcn_wmma_f32_16x16x32_f16(
          false, a, false, bG, (short)0, cz, false, false);
      // D[M][N]: lane holds N=lrow, M=i+8*hs. Store transposed [col][row].
      v8h st;
      #pragma unroll
      for (int i = 0; i < 8; ++i) st[i] = (_Float16)d[i];
      int off = sig * SIG_STRIDE + lrow * COL_STRIDE + chunk * 16 + hs * 8;
      *reinterpret_cast<v8h*>(&myLds[off]) = st;
    }
  }

  __syncthreads();

  // ---------------- column pass: 5 WMMAs --------------------------------
  v8f acc[5];
  #pragma unroll
  for (int sig = 0; sig < 5; ++sig) {
    // B element e <-> K = 16*hs + e ; N = lrow. Contiguous 16 halfs in LDS.
    int off = sig * SIG_STRIDE + lrow * COL_STRIDE + hs * 16;
    v8h b0 = *reinterpret_cast<const v8h*>(&myLds[off]);
    v8h b1 = *reinterpret_cast<const v8h*>(&myLds[off + 8]);
    v16h b;
    #pragma unroll
    for (int i = 0; i < 8; ++i) { b[i] = b0[i]; b[8 + i] = b1[i]; }
    v8f cz = {};
    acc[sig] = __builtin_amdgcn_wmma_f32_16x16x32_f16(
        false, aG, false, b, (short)0, cz, false, false);
  }

  // ---------------- elementwise SSIM + deterministic reduce --------------
  const float C1 = 0.0001f;   // 0.01^2
  const float C2 = 0.0009f;   // 0.03^2
  float lsum = 0.0f;
  #pragma unroll
  for (int i = 0; i < 8; ++i) {
    float mu1 = acc[0][i], mu2 = acc[1][i];
    float e11 = acc[2][i], e22 = acc[3][i], e12 = acc[4][i];
    float mu1sq = mu1 * mu1, mu2sq = mu2 * mu2, mu12 = mu1 * mu2;
    float s1 = e11 - mu1sq, s2 = e22 - mu2sq, s12 = e12 - mu12;
    float num = (2.0f * mu12 + C1) * (2.0f * s12 + C2);
    float den = (mu1sq + mu2sq + C1) * (s1 + s2 + C2);
    lsum += num / den;
  }
  sred[tid] = lsum;
  __syncthreads();
  #pragma unroll
  for (int st = 64; st > 0; st >>= 1) {
    if (tid < st) sred[tid] += sred[tid + st];
    __syncthreads();
  }
  if (tid == 0) blocksum[blockIdx.x] = sred[0];
}

__global__ __launch_bounds__(256) void ssim_finalize_kernel(
    const float* __restrict__ blocksum, float* __restrict__ out, int n)
{
  __shared__ float s[256];
  float acc = 0.0f;
  for (int i = threadIdx.x; i < n; i += 256) acc += blocksum[i];
  s[threadIdx.x] = acc;
  __syncthreads();
  for (int st = 128; st > 0; st >>= 1) {
    if ((int)threadIdx.x < st) s[threadIdx.x] += s[threadIdx.x + st];
    __syncthreads();
  }
  if (threadIdx.x == 0)
    out[0] = 1.0f - s[0] * (1.0f / (float)(NPLANE * IMG_H * IMG_W));
}

extern "C" void kernel_launch(void* const* d_in, const int* in_sizes, int n_in,
                              void* d_out, int out_size, void* d_ws, size_t ws_size,
                              hipStream_t stream) {
  (void)in_sizes; (void)n_in; (void)out_size; (void)ws_size;
  const float* img1 = (const float*)d_in[0];
  const float* img2 = (const float*)d_in[1];
  float* ws  = (float*)d_ws;       // NBLOCKS floats (each written every call)
  float* out = (float*)d_out;

  ssim_main_kernel<<<NBLOCKS, 128, 0, stream>>>(img1, img2, ws);
  ssim_finalize_kernel<<<1, 256, 0, stream>>>(ws, out, NBLOCKS);
}
